// Transformer_66520453480695
// MI455X (gfx1250) — matrix operand order repaired and checked by reading, not compiled, not run on hardware
//
#include <hip/hip_runtime.h>
#include <hip/hip_bf16.h>
#include <math.h>

typedef __attribute__((ext_vector_type(16))) __bf16 v16bf;
typedef __attribute__((ext_vector_type(8)))  float  v8f;

#define LDSS 40  // LDS row stride in bf16 elements (80B, 16B-aligned rows)

// ---------------------------------------------------------------------------
// Embedding gather + positional add: x[b,t,:] = emb_W[tok[b,t],:] + pos_W[t,:]
// ---------------------------------------------------------------------------
__global__ __launch_bounds__(256)
void embed_kernel(const int* __restrict__ tokens, const float* __restrict__ embW,
                  const float* __restrict__ posW, float* __restrict__ x,
                  int T, int D)
{
    const int row = blockIdx.x;           // b*T + t
    const int t   = row % T;
    const int tok = tokens[row];
    const float* er = embW + (size_t)tok * D;
    const float* pr = posW + (size_t)t   * D;
    float* xr = x + (size_t)row * D;
    for (int c = threadIdx.x; c < D; c += 256)
        xr[c] = er[c] + pr[c];
}

// ---------------------------------------------------------------------------
// RMSNorm: y = x * rsqrt(mean(x^2) + 1e-6) * g + be      (one block per row)
// ---------------------------------------------------------------------------
__global__ __launch_bounds__(256)
void rmsnorm_kernel(const float* __restrict__ x, const float* __restrict__ g,
                    const float* __restrict__ be, float* __restrict__ y, int D)
{
    __shared__ float red[8];
    const int row = blockIdx.x;
    const float* xr = x + (size_t)row * D;
    float ss = 0.f;
    for (int c = threadIdx.x; c < D; c += 256) { float v = xr[c]; ss += v * v; }
    #pragma unroll
    for (int off = 16; off > 0; off >>= 1) ss += __shfl_xor(ss, off, 32);
    if ((threadIdx.x & 31) == 0) red[threadIdx.x >> 5] = ss;
    __syncthreads();
    float tot = 0.f;
    #pragma unroll
    for (int w = 0; w < 8; ++w) tot += red[w];
    const float inv = rsqrtf(tot / (float)D + 1e-6f);
    float* yr = y + (size_t)row * D;
    for (int c = threadIdx.x; c < D; c += 256)
        yr[c] = xr[c] * inv * g[c] + be[c];
}

// ---------------------------------------------------------------------------
// bf16 WMMA GEMM: C[M,N] = act(A[M,K] @ W + bias) (+ resid)
//   WT=false: W is [K,N] row-major.   WT=true: W is [N,K] row-major (W^T).
//   Block: 256 threads = 8 waves in 2(M)x4(N) grid.
//   Block tile 128x128, BK=32; each wave owns 4(M)x2(N) 16x16 subtiles
//   -> 8 v_wmma per K-step, 64 accumulator VGPRs per lane.
// ---------------------------------------------------------------------------
template<bool WT, bool ACT_SELU, bool RES>
__global__ __launch_bounds__(256)
void gemm_wmma(const float* __restrict__ A, const float* __restrict__ W,
               const float* __restrict__ bias, const float* __restrict__ resid,
               float* __restrict__ C, int M, int N, int K)
{
    __shared__ __bf16 As[128 * LDSS];
    __shared__ __bf16 Bs[128 * LDSS];   // stored [n][k]
    const int tid  = threadIdx.x;
    const int lane = tid & 31;
    const int wave = tid >> 5;
    const int wm = wave & 1, wn = wave >> 1;    // 2 x 4 wave grid
    const int mbase = blockIdx.y * 128;
    const int nbase = blockIdx.x * 128;
    const int l16 = lane & 15;
    const int hi  = lane >> 4;
    const int ka   = hi * 8;    // A-fragment K interleave
    const int kb16 = hi * 16;   // unused; B fragment also uses ka

    v8f acc[4][2];
    #pragma unroll
    for (int i = 0; i < 4; ++i)
        #pragma unroll
        for (int j = 0; j < 2; ++j)
            acc[i][j] = (v8f){0.f,0.f,0.f,0.f,0.f,0.f,0.f,0.f};

    for (int kb = 0; kb < K; kb += 32) {
        // ---- prefetch next K-slice (lowers to global_prefetch_b8) ----
        if (kb + 32 < K) {
            __builtin_prefetch(&A[(size_t)(mbase + (tid >> 1)) * K + kb + 32 + (tid & 1) * 16], 0, 0);
            if (WT)
                __builtin_prefetch(&W[(size_t)(nbase + (tid >> 1)) * K + kb + 32 + (tid & 1) * 16], 0, 0);
            else
                __builtin_prefetch(&W[(size_t)(kb + 32 + (tid >> 3)) * N + nbase + (tid & 7) * 16], 0, 0);
        }
        // ---- stage A tile 128x32 (float4 coalesced) ----
        #pragma unroll
        for (int j = 0; j < 4; ++j) {
            const int f = tid + j * 256;            // f in [0,1024)
            const int r = f >> 3, c = (f & 7) * 4;
            const float4 v4 = *(const float4*)&A[(size_t)(mbase + r) * K + kb + c];
            As[r * LDSS + c + 0] = (__bf16)v4.x;
            As[r * LDSS + c + 1] = (__bf16)v4.y;
            As[r * LDSS + c + 2] = (__bf16)v4.z;
            As[r * LDSS + c + 3] = (__bf16)v4.w;
        }
        // ---- stage W tile as Bs[n][k] (float4 coalesced in both layouts) ----
        if (WT) {
            #pragma unroll
            for (int j = 0; j < 4; ++j) {
                const int f = tid + j * 256;
                const int n = f >> 3, c = (f & 7) * 4;
                const float4 v4 = *(const float4*)&W[(size_t)(nbase + n) * K + kb + c];
                Bs[n * LDSS + c + 0] = (__bf16)v4.x;
                Bs[n * LDSS + c + 1] = (__bf16)v4.y;
                Bs[n * LDSS + c + 2] = (__bf16)v4.z;
                Bs[n * LDSS + c + 3] = (__bf16)v4.w;
            }
        } else {
            #pragma unroll
            for (int j = 0; j < 4; ++j) {
                const int f = tid + j * 256;
                const int k = f >> 5, n4 = (f & 31) * 4;
                const float4 v4 = *(const float4*)&W[(size_t)(kb + k) * N + nbase + n4];
                Bs[(n4 + 0) * LDSS + k] = (__bf16)v4.x;
                Bs[(n4 + 1) * LDSS + k] = (__bf16)v4.y;
                Bs[(n4 + 2) * LDSS + k] = (__bf16)v4.z;
                Bs[(n4 + 3) * LDSS + k] = (__bf16)v4.w;
            }
        }
        __syncthreads();

        // ---- fragments: 4 A-subtiles, 2 B-subtiles -> 8 WMMAs ----
        v16bf af[4], bf[2];
        #pragma unroll
        for (int i = 0; i < 4; ++i) {
            const __bf16* ap = &As[(wm * 64 + i * 16 + l16) * LDSS];
            #pragma unroll
            for (int e = 0; e < 8; ++e) { af[i][e] = ap[ka + e]; af[i][e + 8] = ap[16 + ka + e]; }
        }
        #pragma unroll
        for (int j = 0; j < 2; ++j) {
            const __bf16* bp = &Bs[(wn * 32 + j * 16 + l16) * LDSS];
            #pragma unroll
            for (int e = 0; e < 8; ++e) { bf[j][e] = bp[ka + e]; bf[j][e + 8] = bp[16 + ka + e]; }
        }
        #pragma unroll
        for (int i = 0; i < 4; ++i)
            #pragma unroll
            for (int j = 0; j < 2; ++j)
                acc[i][j] = __builtin_amdgcn_wmma_f32_16x16x32_bf16(
                    false, af[i], false, bf[j], (short)0, acc[i][j], false, false);
        __syncthreads();
    }

    // ---- epilogue: bias / SELU / residual, f32 store ----
    #pragma unroll
    for (int i = 0; i < 4; ++i) {
        #pragma unroll
        for (int j = 0; j < 2; ++j) {
            const int col = nbase + wn * 32 + j * 16 + l16;
            const float bv = bias ? bias[col] : 0.f;
            #pragma unroll
            for (int r = 0; r < 8; ++r) {
                const int row = mbase + wm * 64 + i * 16 + r + hi * 8;
                float v = acc[i][j][r] + bv;
                if (ACT_SELU) {
                    const float kScale = 1.0507009873554805f;
                    const float kAlpha = 1.6732632423543772f;
                    v = kScale * (v > 0.f ? v : kAlpha * (__expf(v) - 1.f));
                }
                const size_t idx = (size_t)row * N + col;
                if (RES) v += resid[idx];
                C[idx] = v;
            }
        }
    }
}

// ---------------------------------------------------------------------------
// Flash attention (causal, online softmax) with bf16 WMMA.
// One wave per (b, h, 16-row q tile). HD=64, K-chunk = 32.
// qkv layout: [B*T][3*D] with q | k | v sections of D each.
// ---------------------------------------------------------------------------
__global__ __launch_bounds__(32)
void attn_wmma(const float* __restrict__ qkv, float* __restrict__ o)
{
    __shared__ __bf16 P[16 * LDSS];
    const int lane = threadIdx.x & 31;
    const int l16 = lane & 15, hi = lane >> 4;
    const int ka = hi * 8, kb16 = hi * 16;
    const int qt = blockIdx.x, h = blockIdx.y, b = blockIdx.z;
    const int T = 1024, HD = 64, D3 = 3072, D = 1024;
    const float scale = 0.125f;   // 1/sqrt(64)

    // Q fragments (A operand): a0 covers d=0..31, a1 covers d=32..63
    const int tq = qt * 16 + l16;
    const float* qrow = qkv + (size_t)(b * T + tq) * D3 + h * HD;
    v16bf a0, a1;
    #pragma unroll
    for (int e = 0; e < 8; ++e) {
        a0[e] = (__bf16)qrow[ka + e];      a0[e + 8] = (__bf16)qrow[16 + ka + e];
        a1[e] = (__bf16)qrow[32 + ka + e]; a1[e + 8] = (__bf16)qrow[48 + ka + e];
    }

    v8f oa0 = {0,0,0,0,0,0,0,0}, oa1 = {0,0,0,0,0,0,0,0};
    v8f oa2 = {0,0,0,0,0,0,0,0}, oa3 = {0,0,0,0,0,0,0,0};
    float m[8], l[8];
    #pragma unroll
    for (int r = 0; r < 8; ++r) { m[r] = -3.0e38f; l[r] = 0.f; }

    const int kend = qt * 16 + 16;
    for (int kt = 0; kt < kend; kt += 32) {
        // ---- S = Q K^T (two 16x16 N-tiles, contraction over d=64) ----
        v8f s[2];
        #pragma unroll
        for (int nt = 0; nt < 2; ++nt) {
            const int tk = kt + nt * 16 + l16;
            const float* krow = qkv + (size_t)(b * T + tk) * D3 + D + h * HD;
            v16bf b0, b1;
            #pragma unroll
            for (int e = 0; e < 8; ++e) {
                b0[e] = (__bf16)krow[ka + e];      b0[e + 8] = (__bf16)krow[16 + ka + e];
                b1[e] = (__bf16)krow[32 + ka + e]; b1[e + 8] = (__bf16)krow[48 + ka + e];
            }
            v8f z = {0,0,0,0,0,0,0,0};
            z = __builtin_amdgcn_wmma_f32_16x16x32_bf16(false, a0, false, b0, (short)0, z, false, false);
            z = __builtin_amdgcn_wmma_f32_16x16x32_bf16(false, a1, false, b1, (short)0, z, false, false);
            s[nt] = z;
        }
        // ---- scale + causal mask (C layout: row = r + 8*hi, col = l16) ----
        #pragma unroll
        for (int nt = 0; nt < 2; ++nt)
            #pragma unroll
            for (int r = 0; r < 8; ++r) {
                const int rowg = qt * 16 + r + hi * 8;
                const int colg = kt + nt * 16 + l16;
                float v = s[nt][r] * scale;
                if (colg > rowg) v = -3.0e38f;
                s[nt][r] = v;
            }
        // ---- online softmax (row reductions across the 16-lane half) ----
        float alpha[8];
        #pragma unroll
        for (int r = 0; r < 8; ++r) {
            float mx = fmaxf(s[0][r], s[1][r]);
            #pragma unroll
            for (int off = 8; off > 0; off >>= 1) mx = fmaxf(mx, __shfl_xor(mx, off, 32));
            const float mnew = fmaxf(m[r], mx);
            const float p0 = __expf(s[0][r] - mnew);
            const float p1 = __expf(s[1][r] - mnew);
            float rs = p0 + p1;
            #pragma unroll
            for (int off = 8; off > 0; off >>= 1) rs += __shfl_xor(rs, off, 32);
            alpha[r] = __expf(m[r] - mnew);
            l[r] = l[r] * alpha[r] + rs;
            m[r] = mnew;
            s[0][r] = p0; s[1][r] = p1;
        }
        #pragma unroll
        for (int r = 0; r < 8; ++r) {
            oa0[r] *= alpha[r]; oa1[r] *= alpha[r];
            oa2[r] *= alpha[r]; oa3[r] *= alpha[r];
        }
        // ---- re-pack P from C layout to A layout via LDS ----
        __syncthreads();
        #pragma unroll
        for (int r = 0; r < 8; ++r) {
            P[(r + hi * 8) * LDSS + l16]      = (__bf16)s[0][r];
            P[(r + hi * 8) * LDSS + 16 + l16] = (__bf16)s[1][r];
        }
        __syncthreads();
        v16bf pa;
        const __bf16* pr = &P[l16 * LDSS];
        #pragma unroll
        for (int e = 0; e < 8; ++e) { pa[e] = pr[ka + e]; pa[e + 8] = pr[16 + ka + e]; }
        // ---- O += P V (4 N-tiles over HD) ----
        v8f* oacc[4] = { &oa0, &oa1, &oa2, &oa3 };
        #pragma unroll
        for (int nt = 0; nt < 4; ++nt) {
            const int dcol = nt * 16 + l16;
            const float* vbase = qkv + (size_t)(b * T + kt) * D3 + 2 * D + h * HD + dcol;
            v16bf bv;
            #pragma unroll
            for (int e = 0; e < 8; ++e) {
                bv[e] = (__bf16)vbase[(size_t)(ka + e) * D3];
                bv[e + 8] = (__bf16)vbase[(size_t)(16 + ka + e) * D3];
            }
            *oacc[nt] = __builtin_amdgcn_wmma_f32_16x16x32_bf16(false, pa, false, bv,
                                                                (short)0, *oacc[nt], false, false);
        }
    }
    // ---- normalize and write to attention-out buffer [B*T][D] ----
    v8f* oacc[4] = { &oa0, &oa1, &oa2, &oa3 };
    #pragma unroll
    for (int nt = 0; nt < 4; ++nt)
        #pragma unroll
        for (int r = 0; r < 8; ++r) {
            const int rowg = qt * 16 + r + hi * 8;
            const int colg = h * HD + nt * 16 + l16;
            o[(size_t)(b * T + rowg) * D + colg] = (*oacc[nt])[r] / l[r];
        }
}

// ---------------------------------------------------------------------------
// Host-side orchestration
// ---------------------------------------------------------------------------
extern "C" void kernel_launch(void* const* d_in, const int* in_sizes, int n_in,
                              void* d_out, int out_size, void* d_ws, size_t ws_size,
                              hipStream_t stream)
{
    (void)in_sizes; (void)n_in; (void)out_size; (void)ws_size;
    const int B = 2, T = 1024, D = 1024, NH = 16, V = 32000, L = 4;
    const int FF = 4 * D, D3 = 3 * D, M = B * T;

    const int*   tokens = (const int*)  d_in[0];
    const float* embW   = (const float*)d_in[1];
    const float* posW   = (const float*)d_in[2];
    const float* g1     = (const float*)d_in[3];
    const float* be1    = (const float*)d_in[4];
    const float* Wqkv   = (const float*)d_in[5];
    const float* bqkv   = (const float*)d_in[6];
    const float* Wo     = (const float*)d_in[7];
    const float* bo     = (const float*)d_in[8];
    const float* g2     = (const float*)d_in[9];
    const float* be2    = (const float*)d_in[10];
    const float* Wf1    = (const float*)d_in[11];
    const float* bf1    = (const float*)d_in[12];
    const float* Wf2    = (const float*)d_in[13];
    const float* bf2    = (const float*)d_in[14];
    const float* gf     = (const float*)d_in[15];
    const float* bef    = (const float*)d_in[16];
    float* logits = (float*)d_out;

    // workspace carve-out (floats): x | h | qkv | attn-out | ffn-mid  (~80 MB)
    float* x    = (float*)d_ws;
    float* hbuf = x    + (size_t)M * D;
    float* qkv  = hbuf + (size_t)M * D;
    float* obuf = qkv  + (size_t)M * D3;
    float* ffb  = obuf + (size_t)M * D;

    embed_kernel<<<dim3(M), dim3(256), 0, stream>>>(tokens, embW, posW, x, T, D);

    for (int i = 0; i < L; ++i) {
        rmsnorm_kernel<<<dim3(M), dim3(256), 0, stream>>>(x, g1 + (size_t)i * D, be1 + (size_t)i * D, hbuf, D);
        gemm_wmma<false, false, false><<<dim3(D3 / 128, M / 128), dim3(256), 0, stream>>>(
            hbuf, Wqkv + (size_t)i * D * D3, bqkv + (size_t)i * D3, nullptr, qkv, M, D3, D);
        attn_wmma<<<dim3(T / 16, NH, B), dim3(32), 0, stream>>>(qkv, obuf);
        gemm_wmma<false, false, true><<<dim3(D / 128, M / 128), dim3(256), 0, stream>>>(
            obuf, Wo + (size_t)i * D * D, bo + (size_t)i * D, x, x, M, D, D);
        rmsnorm_kernel<<<dim3(M), dim3(256), 0, stream>>>(x, g2 + (size_t)i * D, be2 + (size_t)i * D, hbuf, D);
        gemm_wmma<false, true, false><<<dim3(FF / 128, M / 128), dim3(256), 0, stream>>>(
            hbuf, Wf1 + (size_t)i * D * FF, bf1 + (size_t)i * FF, nullptr, ffb, M, FF, D);
        gemm_wmma<false, false, true><<<dim3(D / 128, M / 128), dim3(256), 0, stream>>>(
            ffb, Wf2 + (size_t)i * FF * D, bf2 + (size_t)i * D, x, x, M, D, FF);
    }

    rmsnorm_kernel<<<dim3(M), dim3(256), 0, stream>>>(x, gf, bef, hbuf, D);
    // tied head: logits = h @ emb_W^T   (emb_W is [V, D] = W^T layout)
    gemm_wmma<true, false, false><<<dim3(V / 128, M / 128), dim3(256), 0, stream>>>(
        hbuf, embW, nullptr, nullptr, logits, M, V, D);
}
